// GRU_Encoder_Block_18502719111443
// MI455X (gfx1250) — compile-verified, matrix operand order checked by reading, not run on hardware
//
#include <hip/hip_runtime.h>
#include <hip/hip_bf16.h>

#define T_STEPS 30
#define BATCH   4096
#define H       128
#define H2      256
#define H3      384

typedef __attribute__((ext_vector_type(16))) __bf16 v16bf;
typedef __attribute__((ext_vector_type(8)))  float  v8f;

union Frag {
    v16bf          v;
    unsigned short s[16];
    uint4          q[2];
};

// Native HW convert (v_cvt_*bf16_f32, RNE) instead of manual bit-twiddling.
__device__ __forceinline__ unsigned short f2bf(float f) {
    union { __bf16 b; unsigned short s; } u;
    u.b = (__bf16)f;
    return u.s;
}
__device__ __forceinline__ float bf2f(unsigned short s) {
    union { unsigned u; float f; } a; a.u = ((unsigned)s) << 16;
    return a.f;
}
__device__ __forceinline__ float sigm(float x) { return 1.0f / (1.0f + __expf(-x)); }

// Load one 16x32 bf16 operand tile (A or B) from row-major memory.
// Per-lane layout matches CDNA5 16-bit WMMA operands:
//   lanes 0-15 : row = lane,    K in {0..7} U {16..23}
//   lanes 16-31: row = lane-16, K in {8..15} U {24..31}
__device__ __forceinline__ Frag ldtile_bf16(const unsigned short* base, int ld, int lane) {
    Frag f;
    int r    = lane & 15;
    int koff = (lane >> 4) << 3;
    const unsigned short* p = base + (size_t)r * ld + koff;
    f.q[0] = *(const uint4*)(p);
    f.q[1] = *(const uint4*)(p + 16);
    return f;
}

// Same gather from an fp32 source (converts to bf16 on the fly).
__device__ __forceinline__ Frag ldtile_f32(const float* base, int ld, int lane) {
    Frag f;
    int r    = lane & 15;
    int koff = (lane >> 4) << 3;
    const float* p = base + (size_t)r * ld + koff;
#pragma unroll
    for (int i = 0; i < 8; ++i) {
        f.s[i]     = f2bf(p[i]);
        f.s[8 + i] = f2bf(p[16 + i]);
    }
    return f;
}

__device__ __forceinline__ v8f wmma_bf16(v8f acc, const Frag& a, const Frag& b) {
    return __builtin_amdgcn_wmma_f32_16x16x32_bf16(false, a.v, false, b.v,
                                                   (short)0, acc, false, false);
}

// ---------------------------------------------------------------------------
// fp32 -> bf16 elementwise convert
// ---------------------------------------------------------------------------
__global__ void cvt_kernel(const float* __restrict__ src,
                           unsigned short* __restrict__ dst, int n) {
    int i = blockIdx.x * blockDim.x + threadIdx.x;
    if (i < n) dst[i] = f2bf(src[i]);
}

// ---------------------------------------------------------------------------
// Input-side GRU GEMM (no recurrence -> fully parallel over all T*B rows):
//   gi = x @ Wih^T + bih   stored bf16 [T*B, 3H]
// ---------------------------------------------------------------------------
__global__ __launch_bounds__(128) void gi_kernel(
    const unsigned short* __restrict__ x,    // [T*B, H] bf16
    const unsigned short* __restrict__ wih,  // [3H, H]  bf16
    const float* __restrict__ bih,
    unsigned short* __restrict__ gi)         // [T*B, 3H] bf16
{
    int lane  = threadIdx.x & 31;
    int w     = threadIdx.x >> 5;
    int row0  = (blockIdx.x * 4 + w) * 16;
    int col   = lane & 15;
    int rbase = (lane >> 4) << 3;

    Frag xa[4];
#pragma unroll
    for (int kk = 0; kk < 4; ++kk)
        xa[kk] = ldtile_bf16(x + (size_t)row0 * H + kk * 32, H, lane);

    for (int j = 0; j < 24; ++j) {           // 24 x 16 = 384 output cols
        v8f a = {};
#pragma unroll
        for (int kk = 0; kk < 4; ++kk) {
            Frag b = ldtile_bf16(wih + (size_t)(j * 16) * H + kk * 32, H, lane);
            a = wmma_bf16(a, xa[kk], b);
        }
        int n = j * 16 + col;
        float bb = bih[n];
#pragma unroll
        for (int p = 0; p < 8; ++p)
            gi[((size_t)row0 + rbase + p) * H3 + n] = f2bf(a[p] + bb);
    }
}

// ---------------------------------------------------------------------------
// GRU recurrence only (gh = h @ Whh^T + bhh, gates, update). Each wave owns
// 16 batch rows; h lives in wave-private LDS (fp32 master for the z*h blend,
// bf16 shadow for direct b128 A-fragment loads). No cross-wave sync.
// rev=1: reads gi / writes out at (T-1-t) => output pre-flipped.
// ---------------------------------------------------------------------------
__global__ __launch_bounds__(128) void gru_rec_kernel(
    const unsigned short* __restrict__ gi,   // [T,B,3H] bf16 (bias folded)
    const float*          __restrict__ h0,   // [B,H]    f32
    const unsigned short* __restrict__ whh,  // [3H,H]   bf16
    const float* __restrict__ bhh,
    unsigned short* __restrict__ out,        // [T,B,H]  bf16
    int rev)
{
    __shared__ __align__(16) float          lds_hf[4][16 * H];
    __shared__ __align__(16) unsigned short lds_hb[4][16 * H];
    int lane  = threadIdx.x & 31;
    int w     = threadIdx.x >> 5;
    int row0  = (blockIdx.x * 4 + w) * 16;
    float*          hf = lds_hf[w];
    unsigned short* hb = lds_hb[w];

    for (int i = lane; i < 16 * H; i += 32) {
        float v = h0[(size_t)row0 * H + i];
        hf[i] = v;
        hb[i] = f2bf(v);
    }
    __syncthreads();

    int col   = lane & 15;
    int rbase = (lane >> 4) << 3;

    for (int t = 0; t < T_STEPS; ++t) {
        int tx = rev ? (T_STEPS - 1 - t) : t;
        const unsigned short* girow = gi + ((size_t)tx * BATCH + row0) * H3;

        // Prefetch next step's gi rows (16 rows x 768B, 6 64B lines per lane)
        // -> global_prefetch_b8; hides HBM latency in the sequential chain.
        if (t + 1 < T_STEPS) {
            int txn = rev ? (tx - 1) : (tx + 1);
            const unsigned short* gn = gi + ((size_t)txn * BATCH + row0) * H3;
            for (int i = lane; i < 16 * 12; i += 32)
                __builtin_prefetch(gn + (i / 12) * H3 + (i % 12) * 32, 0, 3);
        }

        Frag ha[4];
#pragma unroll
        for (int kk = 0; kk < 4; ++kk)
            ha[kk] = ldtile_bf16(hb + kk * 32, H, lane);

        for (int j = 0; j < 8; ++j) {        // 16-col slab of hidden dim
            v8f ar = {}, az = {}, ahn = {};
#pragma unroll
            for (int kk = 0; kk < 4; ++kk) {
                Frag cr = ldtile_bf16(whh + (size_t)(0 * H + j * 16) * H + kk * 32, H, lane);
                Frag cz = ldtile_bf16(whh + (size_t)(1 * H + j * 16) * H + kk * 32, H, lane);
                Frag cn = ldtile_bf16(whh + (size_t)(2 * H + j * 16) * H + kk * 32, H, lane);
                ar  = wmma_bf16(ar,  ha[kk], cr);
                az  = wmma_bf16(az,  ha[kk], cz);
                ahn = wmma_bf16(ahn, ha[kk], cn);
            }
            int n = j * 16 + col;
            float b_hr = bhh[n];
            float b_hz = bhh[H + n];
            float b_hn = bhh[2 * H + n];
#pragma unroll
            for (int p = 0; p < 8; ++p) {
                int   m  = rbase + p;
                const unsigned short* g = girow + (size_t)m * H3;
                float hold = hf[m * H + n];
                float r  = sigm(bf2f(g[n])          + ar[p] + b_hr);
                float z  = sigm(bf2f(g[H + n])      + az[p] + b_hz);
                float nn = tanhf(bf2f(g[2 * H + n]) + r * (ahn[p] + b_hn));
                float hnew = (1.0f - z) * nn + z * hold;
                unsigned short h16 = f2bf(hnew);
                hf[m * H + n] = hnew;
                hb[m * H + n] = h16;
                out[((size_t)tx * BATCH + row0 + m) * H + n] = h16;
            }
        }
    }
}

// ---------------------------------------------------------------------------
// MLP2: relu(LN(LN(x@W1^T+b1)*g1+be1 @ W2^T + b2)*g2+be2) + h   (per-row LN)
// One wave per 16 rows of the flattened [T*B, H] tensor.
// ---------------------------------------------------------------------------
__global__ __launch_bounds__(128) void mlp2_kernel(
    const unsigned short* __restrict__ x,   // [T*B,H] bf16
    const unsigned short* __restrict__ w1,  // [2H,H]  bf16
    const float* __restrict__ b1, const float* __restrict__ g1, const float* __restrict__ be1,
    const unsigned short* __restrict__ w2,  // [H,2H]  bf16
    const float* __restrict__ b2, const float* __restrict__ g2, const float* __restrict__ be2,
    const float* __restrict__ hres,         // [B,H]   f32 residual (broadcast over T)
    unsigned short* __restrict__ out)       // [T*B,H] bf16
{
    __shared__ __align__(16) unsigned short lds_t[4][16 * H2];
    int lane  = threadIdx.x & 31;
    int w     = threadIdx.x >> 5;
    int row0  = (blockIdx.x * 4 + w) * 16;
    unsigned short* tbuf = lds_t[w];
    int col   = lane & 15;
    int rbase = (lane >> 4) << 3;

    Frag xa[4];
#pragma unroll
    for (int kk = 0; kk < 4; ++kk)
        xa[kk] = ldtile_bf16(x + (size_t)row0 * H + kk * 32, H, lane);

    // ---- GEMM1: [16 x 256] = x @ W1^T + b1
    v8f acc[16];
#pragma unroll
    for (int j = 0; j < 16; ++j) {
        v8f a = {};
#pragma unroll
        for (int kk = 0; kk < 4; ++kk) {
            Frag b = ldtile_bf16(w1 + (size_t)(j * 16) * H + kk * 32, H, lane);
            a = wmma_bf16(a, xa[kk], b);
        }
        float bb = b1[j * 16 + col];
#pragma unroll
        for (int p = 0; p < 8; ++p) a[p] += bb;
        acc[j] = a;
    }
    // ---- LayerNorm over 256, write bf16 row-major copy to LDS
#pragma unroll
    for (int p = 0; p < 8; ++p) {
        float s = 0.f, ss = 0.f;
#pragma unroll
        for (int j = 0; j < 16; ++j) { float y = acc[j][p]; s += y; ss += y * y; }
#pragma unroll
        for (int mk = 1; mk <= 8; mk <<= 1) {
            s  += __shfl_xor(s,  mk, 32);
            ss += __shfl_xor(ss, mk, 32);
        }
        float mean = s * (1.0f / H2);
        float var  = ss * (1.0f / H2) - mean * mean;
        float rstd = rsqrtf(var + 1e-5f);
#pragma unroll
        for (int j = 0; j < 16; ++j) {
            int n = j * 16 + col;
            float yn = (acc[j][p] - mean) * rstd * g1[n] + be1[n];
            tbuf[(rbase + p) * H2 + n] = f2bf(yn);
        }
    }
    // ---- GEMM2: [16 x 128] = h1 @ W2^T + b2
    Frag ya[8];
#pragma unroll
    for (int kk = 0; kk < 8; ++kk)
        ya[kk] = ldtile_bf16(tbuf + kk * 32, H2, lane);
    v8f acc2[8];
#pragma unroll
    for (int j = 0; j < 8; ++j) {
        v8f a = {};
#pragma unroll
        for (int kk = 0; kk < 8; ++kk) {
            Frag b = ldtile_bf16(w2 + (size_t)(j * 16) * H2 + kk * 32, H2, lane);
            a = wmma_bf16(a, ya[kk], b);
        }
        float bb = b2[j * 16 + col];
#pragma unroll
        for (int p = 0; p < 8; ++p) a[p] += bb;
        acc2[j] = a;
    }
    int b0 = row0 % BATCH;   // batch index of first row (tiles never cross t)
#pragma unroll
    for (int p = 0; p < 8; ++p) {
        float s = 0.f, ss = 0.f;
#pragma unroll
        for (int j = 0; j < 8; ++j) { float y = acc2[j][p]; s += y; ss += y * y; }
#pragma unroll
        for (int mk = 1; mk <= 8; mk <<= 1) {
            s  += __shfl_xor(s,  mk, 32);
            ss += __shfl_xor(ss, mk, 32);
        }
        float mean = s * (1.0f / H);
        float var  = ss * (1.0f / H) - mean * mean;
        float rstd = rsqrtf(var + 1e-5f);
        int m = rbase + p;
#pragma unroll
        for (int j = 0; j < 8; ++j) {
            int n = j * 16 + col;
            float yn = (acc2[j][p] - mean) * rstd * g2[n] + be2[n];
            float v  = fmaxf(yn, 0.f) + hres[(size_t)(b0 + m) * H + n];
            out[((size_t)row0 + m) * H + n] = f2bf(v);
        }
    }
}

// ---------------------------------------------------------------------------
// Goal projection: relu(LN(goal @ Wg^T + bg)*gg + beg)  -> fp32 h_back0
// ---------------------------------------------------------------------------
__global__ __launch_bounds__(128) void goal_kernel(
    const float* __restrict__ goal,          // [B,H] f32
    const unsigned short* __restrict__ wg,   // [H,H] bf16
    const float* __restrict__ bg, const float* __restrict__ gg, const float* __restrict__ beg,
    float* __restrict__ out)                 // [B,H] f32
{
    int lane  = threadIdx.x & 31;
    int w     = threadIdx.x >> 5;
    int row0  = (blockIdx.x * 4 + w) * 16;
    int col   = lane & 15;
    int rbase = (lane >> 4) << 3;

    Frag xa[4];
#pragma unroll
    for (int kk = 0; kk < 4; ++kk)
        xa[kk] = ldtile_f32(goal + (size_t)row0 * H + kk * 32, H, lane);

    v8f acc[8];
#pragma unroll
    for (int j = 0; j < 8; ++j) {
        v8f a = {};
#pragma unroll
        for (int kk = 0; kk < 4; ++kk) {
            Frag b = ldtile_bf16(wg + (size_t)(j * 16) * H + kk * 32, H, lane);
            a = wmma_bf16(a, xa[kk], b);
        }
        float bb = bg[j * 16 + col];
#pragma unroll
        for (int p = 0; p < 8; ++p) a[p] += bb;
        acc[j] = a;
    }
#pragma unroll
    for (int p = 0; p < 8; ++p) {
        float s = 0.f, ss = 0.f;
#pragma unroll
        for (int j = 0; j < 8; ++j) { float y = acc[j][p]; s += y; ss += y * y; }
#pragma unroll
        for (int mk = 1; mk <= 8; mk <<= 1) {
            s  += __shfl_xor(s,  mk, 32);
            ss += __shfl_xor(ss, mk, 32);
        }
        float mean = s * (1.0f / H);
        float var  = ss * (1.0f / H) - mean * mean;
        float rstd = rsqrtf(var + 1e-5f);
        int m = rbase + p;
#pragma unroll
        for (int j = 0; j < 8; ++j) {
            int n = j * 16 + col;
            float yn = (acc[j][p] - mean) * rstd * gg[n] + beg[n];
            out[((size_t)row0 + m) * H + n] = fmaxf(yn, 0.f);
        }
    }
}

// ---------------------------------------------------------------------------
// Fuse: w = sigmoid(LN(cat(of,ob) @ Wf^T + bf)); out = of*w + ob*(1-w)
// ob is already time-flipped (gru_rec_kernel rev=1 writes it flipped).
// ---------------------------------------------------------------------------
__global__ __launch_bounds__(128) void fuse_kernel(
    const unsigned short* __restrict__ of,   // [T*B,H] bf16
    const unsigned short* __restrict__ ob,   // [T*B,H] bf16 (flipped)
    const unsigned short* __restrict__ wf,   // [H,2H]  bf16
    const float* __restrict__ bf_, const float* __restrict__ gf, const float* __restrict__ bef,
    float* __restrict__ outp)                // [T*B,H] f32
{
    int lane  = threadIdx.x & 31;
    int w     = threadIdx.x >> 5;
    int row0  = (blockIdx.x * 4 + w) * 16;
    int col   = lane & 15;
    int rbase = (lane >> 4) << 3;

    Frag xa[8];
#pragma unroll
    for (int kk = 0; kk < 4; ++kk) {
        xa[kk]     = ldtile_bf16(of + (size_t)row0 * H + kk * 32, H, lane);
        xa[kk + 4] = ldtile_bf16(ob + (size_t)row0 * H + kk * 32, H, lane);
    }
    v8f acc[8];
#pragma unroll
    for (int j = 0; j < 8; ++j) {
        v8f a = {};
#pragma unroll
        for (int kk = 0; kk < 8; ++kk) {
            Frag b = ldtile_bf16(wf + (size_t)(j * 16) * H2 + kk * 32, H2, lane);
            a = wmma_bf16(a, xa[kk], b);
        }
        float bb = bf_[j * 16 + col];
#pragma unroll
        for (int p = 0; p < 8; ++p) a[p] += bb;
        acc[j] = a;
    }
#pragma unroll
    for (int p = 0; p < 8; ++p) {
        float s = 0.f, ss = 0.f;
#pragma unroll
        for (int j = 0; j < 8; ++j) { float y = acc[j][p]; s += y; ss += y * y; }
#pragma unroll
        for (int mk = 1; mk <= 8; mk <<= 1) {
            s  += __shfl_xor(s,  mk, 32);
            ss += __shfl_xor(ss, mk, 32);
        }
        float mean = s * (1.0f / H);
        float var  = ss * (1.0f / H) - mean * mean;
        float rstd = rsqrtf(var + 1e-5f);
        int m = rbase + p;
#pragma unroll
        for (int j = 0; j < 8; ++j) {
            int n  = j * 16 + col;
            float yn  = (acc[j][p] - mean) * rstd * gf[n] + bef[n];
            float wgt = sigm(yn);
            size_t idx = ((size_t)row0 + m) * H + n;
            float fo = bf2f(of[idx]);
            float bo = bf2f(ob[idx]);
            outp[idx] = fo * wgt + bo * (1.0f - wgt);
        }
    }
}

// ---------------------------------------------------------------------------
extern "C" void kernel_launch(void* const* d_in, const int* in_sizes, int n_in,
                              void* d_out, int out_size, void* d_ws, size_t ws_size,
                              hipStream_t stream) {
    (void)in_sizes; (void)n_in; (void)out_size; (void)ws_size;

    const size_t TBH = (size_t)T_STEPS * BATCH * H;

    // ---- workspace layout
    unsigned short* t_bf  = (unsigned short*)d_ws;
    unsigned short* vmid  = t_bf  + TBH;
    unsigned short* vfor  = vmid  + TBH;
    unsigned short* vbac  = vfor  + TBH;
    unsigned short* ofor  = vbac  + TBH;
    unsigned short* oback = ofor  + TBH;
    float*          hb0   = (float*)(oback + TBH);
    unsigned short* gi    = (unsigned short*)(hb0 + (size_t)BATCH * H); // shared gi scratch
    unsigned short* wp    = gi + TBH * 3;
    unsigned short* wih_mid  = wp; wp += H3 * H;
    unsigned short* whh_mid  = wp; wp += H3 * H;
    unsigned short* wih_for  = wp; wp += H3 * H;
    unsigned short* whh_for  = wp; wp += H3 * H;
    unsigned short* wih_back = wp; wp += H3 * H;
    unsigned short* whh_back = wp; wp += H3 * H;
    unsigned short* w1_for   = wp; wp += H2 * H;
    unsigned short* w2_for   = wp; wp += H * H2;
    unsigned short* w1_bac   = wp; wp += H2 * H;
    unsigned short* w2_bac   = wp; wp += H * H2;
    unsigned short* wgoal    = wp; wp += H * H;
    unsigned short* wfuse    = wp; wp += H * H2;

    auto f32 = [&](int i) { return (const float*)d_in[i]; };
    auto cvt = [&](int i, unsigned short* dst, int n) {
        cvt_kernel<<<(n + 255) / 256, 256, 0, stream>>>(f32(i), dst, n);
    };

    // ---- fp32 -> bf16 conversions
    cvt(0,  t_bf,    (int)TBH);
    cvt(4,  wih_mid,  H3 * H);  cvt(5,  whh_mid,  H3 * H);
    cvt(16, wih_for,  H3 * H);  cvt(17, whh_for,  H3 * H);
    cvt(28, wih_back, H3 * H);  cvt(29, whh_back, H3 * H);
    cvt(8,  w1_for,   H2 * H);  cvt(12, w2_for,   H * H2);
    cvt(20, w1_bac,   H2 * H);  cvt(24, w2_bac,   H * H2);
    cvt(32, wgoal,    H * H);   cvt(36, wfuse,    H * H2);

    const int gruBlocks = BATCH / 64;                 // 4 waves x 16 rows
    const int rowBlocks = (int)(TBH / H) / 64;        // T*B rows / 64

    // v_mid = gru_scan(temporal, h[0], gru_mid)
    gi_kernel<<<rowBlocks, 128, 0, stream>>>(t_bf, wih_mid, f32(6), gi);
    gru_rec_kernel<<<gruBlocks, 128, 0, stream>>>(gi, f32(2), whh_mid, f32(7), vmid, 0);
    // v_for = h + mlp2(v_mid, mlp_for)
    mlp2_kernel<<<rowBlocks, 128, 0, stream>>>(vmid, w1_for, f32(9), f32(10), f32(11),
                                               w2_for, f32(13), f32(14), f32(15),
                                               f32(2), vfor);
    // out_for = gru_scan(v_for, h_temp, gru_for)
    gi_kernel<<<rowBlocks, 128, 0, stream>>>(vfor, wih_for, f32(18), gi);
    gru_rec_kernel<<<gruBlocks, 128, 0, stream>>>(gi, f32(1), whh_for, f32(19), ofor, 0);
    // v_back = h + mlp2(v_mid, mlp_bac)
    mlp2_kernel<<<rowBlocks, 128, 0, stream>>>(vmid, w1_bac, f32(21), f32(22), f32(23),
                                               w2_bac, f32(25), f32(26), f32(27),
                                               f32(2), vbac);
    // h_back0 = relu(LN(goal @ Wg^T))
    goal_kernel<<<BATCH / 64, 128, 0, stream>>>(f32(3), wgoal, f32(33), f32(34), f32(35), hb0);
    // out_back_flip = flip(gru_scan(flip(v_back), h_back0, gru_back))
    gi_kernel<<<rowBlocks, 128, 0, stream>>>(vbac, wih_back, f32(30), gi);
    gru_rec_kernel<<<gruBlocks, 128, 0, stream>>>(gi, hb0, whh_back, f32(31), oback, 1);
    // fuse + blend -> d_out (fp32)
    fuse_kernel<<<rowBlocks, 128, 0, stream>>>(ofor, oback, wfuse,
                                               f32(37), f32(38), f32(39), (float*)d_out);
}
